// SEEDDecoderLayer_24464133718853
// MI455X (gfx1250) — matrix operand order repaired and checked by reading, not compiled, not run on hardware
//
#include <hip/hip_runtime.h>

// ---------------------------------------------------------------------------
// CDNA5 (gfx1250, wave32) WMMA types and fragment helpers
// ---------------------------------------------------------------------------
typedef __attribute__((ext_vector_type(16))) _Float16 v16h;
typedef __attribute__((ext_vector_type(8)))  float    v8f;
typedef __attribute__((ext_vector_type(4)))  float    v4f;

union F16Frag {
  v16h     v;
  v4f      f4[2];
  _Float16 h[16];
};

constexpr int kB  = 4;
constexpr int kLQ = 1000;
constexpr int kD  = 256;
constexpr int kNH = 8;
constexpr int kDH = 32;
constexpr int kHW = 188;
constexpr int kLV = kHW * kHW;   // 35344
constexpr int kNP = 25;

// A fragment: 16x32 f16, lane = M row (lane&15); half-wave hg=lane>>4:
// halves 0..7 -> K = hg*8 + e ; halves 8..15 -> K = 16 + hg*8 + e.
__device__ inline v16h load_a_frag(const _Float16* __restrict__ tile, int ld, int lane) {
  int row = lane & 15;
  int hg  = (lane >> 4) & 1;
  const _Float16* p = tile + (long)row * ld + hg * 8;
  F16Frag u;
  u.f4[0] = *(const v4f*)(p);
  u.f4[1] = *(const v4f*)(p + 16);
  return u.v;
}

// B fragment: 32x16 f16, lane = N col (lane&15); halves 0..7 -> K = hg*8+e ; halves 8..15 -> K = 16 + hg*8 + e.
// wrow must point at W[col][k0] (K-contiguous row of the N x K weight matrix).
__device__ inline v16h load_b_frag(const _Float16* __restrict__ wrow, int hg) {
  const _Float16* p = wrow + hg * 8;
  F16Frag u;
  u.f4[0] = *(const v4f*)(p);
  u.f4[1] = *(const v4f*)(p + 16);
  return u.v;
}

// ---------------------------------------------------------------------------
// f32 -> f16 convert with optional K padding (zero fill)
// ---------------------------------------------------------------------------
__global__ void __launch_bounds__(256)
cvt_pad_f16(const float* __restrict__ src, _Float16* __restrict__ dst,
            long M, int K, int Kp) {
  long i = (long)blockIdx.x * blockDim.x + threadIdx.x;
  if (i >= M * (long)Kp) return;
  long r = i / Kp;
  int  c = (int)(i - r * Kp);
  dst[i] = (c < K) ? (_Float16)src[r * (long)K + c] : (_Float16)0.0f;
}

// ---------------------------------------------------------------------------
// Generic GEMM: C[M,N] = act(A[M,Kp] @ W[N,Kp]^T + bias), f16 in, f32 acc.
// One wave computes a 32x32 C macro-tile (2x2 wmma): each A fragment is
// reused across 2 N-tiles and each B fragment across 2 M-tiles, so the
// steady-state loop costs 8 b128 loads per 4 v_wmma (2 loads/wmma).
// M must be a multiple of 32. Full-width N tiles take a guard-free loop
// (no per-iteration EXEC save/restore); ragged tails take the masked loop.
// ---------------------------------------------------------------------------
__global__ void __launch_bounds__(256)
gemm_f16_wmma(const _Float16* __restrict__ A, int lda,
              const _Float16* __restrict__ W, int ldw,
              const float* __restrict__ bias,
              float* __restrict__ Cf, _Float16* __restrict__ Ch,
              int M, int N, int Kp, int relu) {
  int lane = threadIdx.x & 31;
  int wave = threadIdx.x >> 5;
  int m0 = blockIdx.x * 32;
  int n0 = (blockIdx.y * 8 + wave) * 32;
  if (n0 >= N) return;

  int ln16 = lane & 15;
  int hg   = (lane >> 4) & 1;
  int c0 = n0 + ln16;
  int c1 = n0 + 16 + ln16;

  const _Float16* a0 = A + (long)m0 * lda;
  const _Float16* a1 = A + (long)(m0 + 16) * lda;
  const _Float16* w0 = W + (long)c0 * ldw;
  const _Float16* w1 = W + (long)c1 * ldw;

  v8f acc00 = {}, acc01 = {}, acc10 = {}, acc11 = {};

  if (n0 + 32 <= N) {
    // ---- full tile: no lane guards in the hot loop ----
    for (int k0 = 0; k0 < Kp; k0 += 32) {
      if (k0 + 32 < Kp) {
        __builtin_prefetch(a0 + k0 + 32, 0, 3);   // global_prefetch_b8 (near)
        __builtin_prefetch(a1 + k0 + 32, 0, 3);
      }
      v16h af0 = load_a_frag(a0 + k0, lda, lane);
      v16h af1 = load_a_frag(a1 + k0, lda, lane);
      v16h bf0 = load_b_frag(w0 + k0, hg);
      v16h bf1 = load_b_frag(w1 + k0, hg);
      acc00 = __builtin_amdgcn_wmma_f32_16x16x32_f16(false, af0, false, bf0,
                                                     (short)0, acc00, false, false);
      acc01 = __builtin_amdgcn_wmma_f32_16x16x32_f16(false, af0, false, bf1,
                                                     (short)0, acc01, false, false);
      acc10 = __builtin_amdgcn_wmma_f32_16x16x32_f16(false, af1, false, bf0,
                                                     (short)0, acc10, false, false);
      acc11 = __builtin_amdgcn_wmma_f32_16x16x32_f16(false, af1, false, bf1,
                                                     (short)0, acc11, false, false);
    }
  } else {
    // ---- ragged N tail (heads with N=200/40) ----
    v4f z4 = {};
    for (int k0 = 0; k0 < Kp; k0 += 32) {
      v16h af0 = load_a_frag(a0 + k0, lda, lane);
      v16h af1 = load_a_frag(a1 + k0, lda, lane);
      F16Frag b0f, b1f;
      if (c0 < N) b0f.v = load_b_frag(w0 + k0, hg);
      else        { b0f.f4[0] = z4; b0f.f4[1] = z4; }
      if (c1 < N) b1f.v = load_b_frag(w1 + k0, hg);
      else        { b1f.f4[0] = z4; b1f.f4[1] = z4; }
      acc00 = __builtin_amdgcn_wmma_f32_16x16x32_f16(false, af0, false, b0f.v,
                                                     (short)0, acc00, false, false);
      acc01 = __builtin_amdgcn_wmma_f32_16x16x32_f16(false, af0, false, b1f.v,
                                                     (short)0, acc01, false, false);
      acc10 = __builtin_amdgcn_wmma_f32_16x16x32_f16(false, af1, false, b0f.v,
                                                     (short)0, acc10, false, false);
      acc11 = __builtin_amdgcn_wmma_f32_16x16x32_f16(false, af1, false, b1f.v,
                                                     (short)0, acc11, false, false);
    }
  }

  // ---- epilogue: bias + activation + store (f32 and/or f16) ----
  float bv0 = (bias && c0 < N) ? bias[c0] : 0.0f;
  float bv1 = (bias && c1 < N) ? bias[c1] : 0.0f;
  int rbase0 = m0 + hg * 8;
  int rbase1 = m0 + 16 + hg * 8;
#pragma unroll
  for (int r = 0; r < 8; ++r) {
    long row0 = rbase0 + r;
    long row1 = rbase1 + r;
    float v00 = acc00[r] + bv0, v01 = acc01[r] + bv1;
    float v10 = acc10[r] + bv0, v11 = acc11[r] + bv1;
    if (relu) {
      v00 = fmaxf(v00, 0.0f); v01 = fmaxf(v01, 0.0f);
      v10 = fmaxf(v10, 0.0f); v11 = fmaxf(v11, 0.0f);
    }
    if (c0 < N) {
      if (Cf) { Cf[row0 * N + c0] = v00; Cf[row1 * N + c0] = v10; }
      if (Ch) { Ch[row0 * N + c0] = (_Float16)v00; Ch[row1 * N + c0] = (_Float16)v10; }
    }
    if (c1 < N) {
      if (Cf) { Cf[row0 * N + c1] = v01; Cf[row1 * N + c1] = v11; }
      if (Ch) { Ch[row0 * N + c1] = (_Float16)v01; Ch[row1 * N + c1] = (_Float16)v11; }
    }
  }
}

// ---------------------------------------------------------------------------
// Flash attention: one wave per (b, h, 16-query tile); DH=32 so QK^T is one
// wmma per 16-key tile; P*V is 2 wmmas per 32-key chunk with an LDS transpose
// of the P tile (C layout -> A layout). Online softmax in registers.
// ---------------------------------------------------------------------------
__global__ void __launch_bounds__(128)
attn_flash(const _Float16* __restrict__ qkv,   // (B,LQ,768): [q|k|v] per row
           _Float16* __restrict__ sa16) {      // (B,LQ,256)
  __shared__ __align__(16) _Float16 ldsP[4][16 * 32];
  int lane = threadIdx.x & 31;
  int wave = threadIdx.x >> 5;
  int qt = blockIdx.x * 4 + wave;
  int h  = blockIdx.y;
  int b  = blockIdx.z;
  int q0 = qt * 16;
  if (q0 >= kLQ) return;

  int hg   = (lane >> 4) & 1;
  int ln16 = lane & 15;

  // Q fragment, pre-scaled by 1/sqrt(DH)
  int qrow = q0 + ln16;
  if (qrow >= kLQ) qrow = kLQ - 1;
  const _Float16* qp = qkv + ((long)(b * kLQ + qrow)) * 768 + h * kDH + hg * 8;
  F16Frag qa;
  qa.f4[0] = *(const v4f*)(qp);
  qa.f4[1] = *(const v4f*)(qp + 16);
#pragma unroll
  for (int e = 0; e < 16; ++e)
    qa.h[e] = (_Float16)((float)qa.h[e] * 0.17677669529663687f);

  const _Float16* kbase = qkv + ((long)b * kLQ) * 768 + 256 + h * kDH;
  const _Float16* vbase = qkv + ((long)b * kLQ) * 768 + 512 + h * kDH;

  float m[8], l[8];
#pragma unroll
  for (int r = 0; r < 8; ++r) { m[r] = -3.0e38f; l[r] = 0.0f; }
  v8f o0 = {};
  v8f o1 = {};
  _Float16* lp = &ldsP[wave][0];
  v4f z4 = {};

  for (int kc = 0; kc < 32; ++kc) {       // 32 chunks x 32 keys covers 1000
    int kk = kc * 32;
    v8f s0, s1;
    {
      int key = kk + ln16;
      F16Frag kb;
      if (key < kLQ) kb.v = load_b_frag(kbase + (long)key * 768, hg);
      else           { kb.f4[0] = z4; kb.f4[1] = z4; }
      v8f z = {};
      s0 = __builtin_amdgcn_wmma_f32_16x16x32_f16(false, qa.v, false, kb.v,
                                                  (short)0, z, false, false);
      if (key >= kLQ) {
#pragma unroll
        for (int r = 0; r < 8; ++r) s0[r] = -3.0e38f;
      }
    }
    {
      int key = kk + 16 + ln16;
      F16Frag kb;
      if (key < kLQ) kb.v = load_b_frag(kbase + (long)key * 768, hg);
      else           { kb.f4[0] = z4; kb.f4[1] = z4; }
      v8f z = {};
      s1 = __builtin_amdgcn_wmma_f32_16x16x32_f16(false, qa.v, false, kb.v,
                                                  (short)0, z, false, false);
      if (key >= kLQ) {
#pragma unroll
        for (int r = 0; r < 8; ++r) s1[r] = -3.0e38f;
      }
    }

    // online softmax over this 32-key chunk (row reductions within 16-lane group)
#pragma unroll
    for (int r = 0; r < 8; ++r) {
      float v = fmaxf(s0[r], s1[r]);
      v = fmaxf(v, __shfl_xor(v, 1, 32));
      v = fmaxf(v, __shfl_xor(v, 2, 32));
      v = fmaxf(v, __shfl_xor(v, 4, 32));
      v = fmaxf(v, __shfl_xor(v, 8, 32));
      float mnew = fmaxf(m[r], v);
      float f = __expf(m[r] - mnew);
      m[r] = mnew;
      float p0 = __expf(s0[r] - mnew);
      float p1 = __expf(s1[r] - mnew);
      float ps = p0 + p1;
      ps += __shfl_xor(ps, 1, 32);
      ps += __shfl_xor(ps, 2, 32);
      ps += __shfl_xor(ps, 4, 32);
      ps += __shfl_xor(ps, 8, 32);
      l[r] = l[r] * f + ps;
      o0[r] *= f;
      o1[r] *= f;
      int prow = hg * 8 + r;
      lp[prow * 32 + ln16]      = (_Float16)p0;   // C layout -> LDS row-major
      lp[prow * 32 + 16 + ln16] = (_Float16)p1;
    }
    asm volatile("s_wait_dscnt 0" ::: "memory");  // wave-private LDS RAW

    // P tile re-read in A layout
    F16Frag pa;
    {
      const _Float16* pr = lp + ln16 * 32 + hg * 8;
      pa.f4[0] = *(const v4f*)(pr);
      pa.f4[1] = *(const v4f*)(pr + 16);
    }
    // V fragments: B is 32keys x 32dh, two 16-wide N tiles
    F16Frag vf0, vf1;
#pragma unroll
    for (int e = 0; e < 16; ++e) {
      int k = kk + hg * 8 + (e < 8 ? e : e + 8);
      bool ok = (k < kLQ);
      vf0.h[e] = ok ? vbase[(long)k * 768 + ln16]      : (_Float16)0.0f;
      vf1.h[e] = ok ? vbase[(long)k * 768 + 16 + ln16] : (_Float16)0.0f;
    }
    o0 = __builtin_amdgcn_wmma_f32_16x16x32_f16(false, pa.v, false, vf0.v,
                                                (short)0, o0, false, false);
    o1 = __builtin_amdgcn_wmma_f32_16x16x32_f16(false, pa.v, false, vf1.v,
                                                (short)0, o1, false, false);
  }

#pragma unroll
  for (int r = 0; r < 8; ++r) {
    int row = q0 + hg * 8 + r;
    if (row < kLQ) {
      float inv = 1.0f / l[r];
      long o = ((long)(b * kLQ + row)) * kD + h * kDH;
      sa16[o + ln16]      = (_Float16)(o0[r] * inv);
      sa16[o + 16 + ln16] = (_Float16)(o1[r] * inv);
    }
  }
}

// ---------------------------------------------------------------------------
// y = LayerNorm(x + res) * g + b ; one 256-thread block per row (D=256)
// ---------------------------------------------------------------------------
__global__ void __launch_bounds__(256)
residual_ln(const float* __restrict__ x, const float* __restrict__ res,
            const float* __restrict__ g, const float* __restrict__ bta,
            float* __restrict__ outf, _Float16* __restrict__ outh) {
  __shared__ float red[256];
  int row = blockIdx.x;
  int t = threadIdx.x;
  float v = x[(long)row * kD + t] + (res ? res[(long)row * kD + t] : 0.0f);
  red[t] = v;
  __syncthreads();
  for (int s = 128; s > 0; s >>= 1) {
    if (t < s) red[t] += red[t + s];
    __syncthreads();
  }
  float mean = red[0] * (1.0f / kD);
  __syncthreads();
  float d = v - mean;
  red[t] = d * d;
  __syncthreads();
  for (int s = 128; s > 0; s >>= 1) {
    if (t < s) red[t] += red[t + s];
    __syncthreads();
  }
  float var = red[0] * (1.0f / kD);
  float y = d * rsqrtf(var + 1e-5f) * g[t] + bta[t];
  if (outf) outf[(long)row * kD + t] = y;
  if (outh) outh[(long)row * kD + t] = (_Float16)y;
}

// ---------------------------------------------------------------------------
// Row softmax over 25 elements; one wave per row
// ---------------------------------------------------------------------------
__global__ void __launch_bounds__(256)
softmax25(const float* __restrict__ in, float* __restrict__ out, long rows) {
  int lane = threadIdx.x & 31;
  int wave = threadIdx.x >> 5;
  long row = (long)blockIdx.x * 8 + wave;
  if (row >= rows) return;
  const float* p = in + row * kNP;
  float v = (lane < kNP) ? p[lane] : -3.4e38f;
  float mx = v;
  mx = fmaxf(mx, __shfl_xor(mx, 1, 32));
  mx = fmaxf(mx, __shfl_xor(mx, 2, 32));
  mx = fmaxf(mx, __shfl_xor(mx, 4, 32));
  mx = fmaxf(mx, __shfl_xor(mx, 8, 32));
  mx = fmaxf(mx, __shfl_xor(mx, 16, 32));
  float e = (lane < kNP) ? __expf(v - mx) : 0.0f;
  float s = e;
  s += __shfl_xor(s, 1, 32);
  s += __shfl_xor(s, 2, 32);
  s += __shfl_xor(s, 4, 32);
  s += __shfl_xor(s, 8, 32);
  s += __shfl_xor(s, 16, 32);
  if (lane < kNP) out[row * kNP + lane] = e / s;
}

// ---------------------------------------------------------------------------
// Build sampling grid per (b,q,h,p): rotated kernel offsets + learned deform
// ---------------------------------------------------------------------------
__global__ void __launch_bounds__(256)
build_grid(const float* __restrict__ refw,   // (B,LQ,10)
           const float* __restrict__ boxoff, // (B,LQ,40)  [h*5 + j]
           const float* __restrict__ deform, // (B,LQ,400) [h*50 + p*2 + c]
           const float* __restrict__ kidx,   // (25,2)
           float* __restrict__ grid) {       // (B,LQ,NH,25,2)
  int t = blockIdx.x * 256 + threadIdx.x;
  if (t >= kB * kLQ * kNH * kNP) return;
  int p  = t % kNP;
  int h  = (t / kNP) % kNH;
  int bq = t / (kNP * kNH);
  const float* rw = refw + (long)bq * 10;
  const float* bo = boxoff + (long)bq * 40 + h * 5;
  float ang = (rw[6] + bo[4] * (1.0f / 16.0f)) * 6.283185307179586f;
  float cx = rw[0] + bo[0] * 0.125f * rw[3];
  float cy = rw[1] + bo[1] * 0.125f * rw[4];
  float sw = rw[3] + bo[2] * 0.125f * rw[3];
  float sh = rw[4] + bo[3] * 0.125f * rw[4];
  float ca = cosf(ang), sa = sinf(ang);
  float fx = kidx[p * 2 + 0] * fmaxf(sw, 0.0f);
  float fy = kidx[p * 2 + 1] * fmaxf(sh, 0.0f);
  float gx = cx + fx * ca - fy * sa;
  float gy = cy + fx * sa + fy * ca;
  gx += deform[(long)bq * 400 + h * 50 + p * 2 + 0] * (1.0f / 188.0f);
  gy += deform[(long)bq * 400 + h * 50 + p * 2 + 1] * (1.0f / 188.0f);
  long o = (((long)bq * kNH + h) * kNP + p) * 2;
  grid[o + 0] = gx;
  grid[o + 1] = gy;
}

// ---------------------------------------------------------------------------
// Deformable bilinear sampling + attention-weighted reduce.
// One wave per (b,q,h); lane = channel d (DH=32 maps exactly onto wave32).
// ---------------------------------------------------------------------------
__global__ void __launch_bounds__(256)
deform_sample(const _Float16* __restrict__ val,  // (B, LV, 256)
              const float* __restrict__ grid,    // (B,LQ,NH,25,2)
              const float* __restrict__ aw,      // (B,LQ,NH,25)
              _Float16* __restrict__ outh) {     // (B,LQ,256)
  int lane = threadIdx.x & 31;
  int wave = threadIdx.x >> 5;
  int idx = blockIdx.x * 8 + wave;      // (b*LQ + q)*NH + h, 0..31999
  int h  = idx % kNH;
  int bq = idx / kNH;
  int b  = bq / kLQ;
  const float* gp = grid + (long)idx * kNP * 2;
  const float* ap = aw + (long)idx * kNP;
  const _Float16* vb = val + (long)b * kLV * kD + h * kDH + lane;
  float acc = 0.0f;
  for (int p = 0; p < kNP; ++p) {
    float x = gp[p * 2 + 0] * (float)kHW - 0.5f;
    float y = gp[p * 2 + 1] * (float)kHW - 0.5f;
    float x0 = floorf(x), y0 = floorf(y);
    float fx = x - x0, fy = y - y0;
    int xi = (int)x0, yi = (int)y0;
    float s = 0.0f;
    if (xi >= 0 && xi < kHW && yi >= 0 && yi < kHW)
      s += (1.0f - fx) * (1.0f - fy) * (float)vb[((long)yi * kHW + xi) * kD];
    if (xi + 1 >= 0 && xi + 1 < kHW && yi >= 0 && yi < kHW)
      s += fx * (1.0f - fy) * (float)vb[((long)yi * kHW + xi + 1) * kD];
    if (xi >= 0 && xi < kHW && yi + 1 >= 0 && yi + 1 < kHW)
      s += (1.0f - fx) * fy * (float)vb[((long)(yi + 1) * kHW + xi) * kD];
    if (xi + 1 >= 0 && xi + 1 < kHW && yi + 1 >= 0 && yi + 1 < kHW)
      s += fx * fy * (float)vb[((long)(yi + 1) * kHW + xi + 1) * kD];
    acc += ap[p] * s;
  }
  outh[(long)bq * kD + h * kDH + lane] = (_Float16)acc;
}

// ---------------------------------------------------------------------------
// Host-side orchestration
// ---------------------------------------------------------------------------
extern "C" void kernel_launch(void* const* d_in, const int* in_sizes, int n_in,
                              void* d_out, int out_size, void* d_ws, size_t ws_size,
                              hipStream_t stream) {
  (void)in_sizes; (void)n_in; (void)out_size; (void)ws_size;

  const float* memory   = (const float*)d_in[1];
  const float* ref_w    = (const float*)d_in[4];
  const float* kidx     = (const float*)d_in[5];
  const float* pe_w0    = (const float*)d_in[6];
  const float* pe_b0    = (const float*)d_in[7];
  const float* pe_w1    = (const float*)d_in[8];
  const float* pe_b1    = (const float*)d_in[9];
  const float* pe_w2    = (const float*)d_in[10];
  const float* pe_b2    = (const float*)d_in[11];
  const float* sa_in_w  = (const float*)d_in[12];
  const float* sa_in_b  = (const float*)d_in[13];
  const float* sa_out_w = (const float*)d_in[14];
  const float* sa_out_b = (const float*)d_in[15];
  const float* ca_val_w = (const float*)d_in[16];
  const float* ca_val_b = (const float*)d_in[17];
  const float* ca_att_w = (const float*)d_in[18];
  const float* ca_att_b = (const float*)d_in[19];
  const float* ca_box_w = (const float*)d_in[20];
  const float* ca_box_b = (const float*)d_in[21];
  const float* ca_off_w = (const float*)d_in[22];
  const float* ca_off_b = (const float*)d_in[23];
  const float* ca_out_w = (const float*)d_in[24];
  const float* ca_out_b = (const float*)d_in[25];
  const float* ff_w1    = (const float*)d_in[26];
  const float* ff_b1    = (const float*)d_in[27];
  const float* ff_w2    = (const float*)d_in[28];
  const float* ff_b2    = (const float*)d_in[29];
  const float* ln1_g    = (const float*)d_in[30];
  const float* ln1_b    = (const float*)d_in[31];
  const float* ln2_g    = (const float*)d_in[32];
  const float* ln2_b    = (const float*)d_in[33];
  const float* ln3_g    = (const float*)d_in[34];
  const float* ln3_b    = (const float*)d_in[35];

  char* base = (char*)d_ws;
  size_t off = 0;
  auto alloc = [&](size_t bytes) -> void* {
    off = (off + 255) & ~(size_t)255;
    void* p = base + off;
    off += bytes;
    return p;
  };
  auto af16 = [&](size_t n) { return (_Float16*)alloc(n * sizeof(_Float16)); };
  auto af32 = [&](size_t n) { return (float*)alloc(n * sizeof(float)); };

  const long M = (long)kB * kLQ;          // 4000 rows (multiple of 32)
  const long MV = (long)kB * kLV;         // 141376 rows (multiple of 32)

  // f16 weight copies (converted once)
  _Float16* w_pe0  = af16(256 * 32);      // K=10 padded to 32
  _Float16* w_pe1  = af16(256 * 256);
  _Float16* w_pe2  = af16(256 * 256);
  _Float16* w_sain = af16(768 * 256);
  _Float16* w_saou = af16(256 * 256);
  _Float16* w_cav  = af16(256 * 256);
  _Float16* w_caa  = af16(200 * 256);
  _Float16* w_cab  = af16(40 * 256);
  _Float16* w_cao  = af16(400 * 256);
  _Float16* w_caot = af16(256 * 256);
  _Float16* w_ff1  = af16(1024 * 256);
  _Float16* w_ff2  = af16(256 * 1024);

  // activations
  _Float16* rw16    = af16(M * 32);
  _Float16* h0_16   = af16(M * 256);
  _Float16* h1_16   = af16(M * 256);
  float*    q0_f    = af32(M * 256);
  _Float16* q0_16   = af16(M * 256);
  _Float16* qkv16   = af16(M * 768);
  _Float16* sa16    = af16(M * 256);
  float*    saout_f = af32(M * 256);
  float*    q1_f    = af32(M * 256);
  _Float16* q1_16   = af16(M * 256);
  float*    awraw   = af32(M * 200);
  float*    awsoft  = af32(M * 200);
  float*    boxoff  = af32(M * 40);
  float*    deform  = af32(M * 400);
  float*    gridb   = af32(M * kNH * kNP * 2);
  _Float16* mem16   = af16(MV * 256);
  _Float16* val16   = af16(MV * 256);
  _Float16* ca16    = af16(M * 256);
  float*    caout_f = af32(M * 256);
  float*    q2_f    = af32(M * 256);
  _Float16* q2_16   = af16(M * 256);
  _Float16* ffh16   = af16(M * 1024);
  float*    ffo_f   = af32(M * 256);

  auto cvt = [&](const float* s, _Float16* d, long Mr, int K, int Kp) {
    long total = Mr * (long)Kp;
    unsigned blocks = (unsigned)((total + 255) / 256);
    cvt_pad_f16<<<dim3(blocks), dim3(256), 0, stream>>>(s, d, Mr, K, Kp);
  };
  auto gemm = [&](const _Float16* A, int lda, const _Float16* W, int ldw,
                  const float* bias, float* Cf, _Float16* Ch,
                  int Mr, int N, int Kp, int relu) {
    int mt  = Mr / 32;                 // 32-row macro tiles
    int ntw = (N + 31) / 32;           // 32-col macro tiles (1 per wave)
    int gy  = (ntw + 7) / 8;
    gemm_f16_wmma<<<dim3(mt, gy), dim3(256), 0, stream>>>(
        A, lda, W, ldw, bias, Cf, Ch, Mr, N, Kp, relu);
  };

  // --- weight conversions ---
  cvt(pe_w0, w_pe0, 256, 10, 32);
  cvt(pe_w1, w_pe1, 256, 256, 256);
  cvt(pe_w2, w_pe2, 256, 256, 256);
  cvt(sa_in_w, w_sain, 768, 256, 256);
  cvt(sa_out_w, w_saou, 256, 256, 256);
  cvt(ca_val_w, w_cav, 256, 256, 256);
  cvt(ca_att_w, w_caa, 200, 256, 256);
  cvt(ca_box_w, w_cab, 40, 256, 256);
  cvt(ca_off_w, w_cao, 400, 256, 256);
  cvt(ca_out_w, w_caot, 256, 256, 256);
  cvt(ff_w1, w_ff1, 1024, 256, 256);
  cvt(ff_w2, w_ff2, 256, 1024, 1024);

  // --- PE MLP: query0 = W2(relu(W1(relu(W0 ref)))) ---
  cvt(ref_w, rw16, M, 10, 32);
  gemm(rw16, 32, w_pe0, 32, pe_b0, nullptr, h0_16, (int)M, 256, 32, 1);
  gemm(h0_16, 256, w_pe1, 256, pe_b1, nullptr, h1_16, (int)M, 256, 256, 1);
  gemm(h1_16, 256, w_pe2, 256, pe_b2, q0_f, q0_16, (int)M, 256, 256, 0);

  // --- self-attention ---
  gemm(q0_16, 256, w_sain, 256, sa_in_b, nullptr, qkv16, (int)M, 768, 256, 0);
  attn_flash<<<dim3(16, kNH, kB), dim3(128), 0, stream>>>(qkv16, sa16);
  gemm(sa16, 256, w_saou, 256, sa_out_b, saout_f, nullptr, (int)M, 256, 256, 0);
  residual_ln<<<dim3((unsigned)M), dim3(256), 0, stream>>>(
      saout_f, q0_f, ln1_g, ln1_b, q1_f, q1_16);

  // --- cross-attention value projection (dominant, bandwidth-bound GEMM) ---
  cvt(memory, mem16, MV, 256, 256);
  gemm(mem16, 256, w_cav, 256, ca_val_b, nullptr, val16, (int)MV, 256, 256, 0);

  // --- attention weights / box / deform heads ---
  gemm(q1_16, 256, w_caa, 256, ca_att_b, awraw, nullptr, (int)M, 200, 256, 0);
  softmax25<<<dim3((unsigned)M), dim3(256), 0, stream>>>(awraw, awsoft, M * kNH);
  gemm(q1_16, 256, w_cab, 256, ca_box_b, boxoff, nullptr, (int)M, 40, 256, 0);
  gemm(q1_16, 256, w_cao, 256, ca_off_b, deform, nullptr, (int)M, 400, 256, 0);
  {
    long total = M * kNH * kNP;
    build_grid<<<dim3((unsigned)((total + 255) / 256)), dim3(256), 0, stream>>>(
        ref_w, boxoff, deform, kidx, gridb);
  }
  deform_sample<<<dim3((unsigned)M), dim3(256), 0, stream>>>(
      val16, gridb, awsoft, ca16);
  gemm(ca16, 256, w_caot, 256, ca_out_b, caout_f, nullptr, (int)M, 256, 256, 0);
  residual_ln<<<dim3((unsigned)M), dim3(256), 0, stream>>>(
      caout_f, q1_f, ln2_g, ln2_b, q2_f, q2_16);

  // --- FFN ---
  gemm(q2_16, 256, w_ff1, 256, ff_b1, nullptr, ffh16, (int)M, 1024, 256, 1);
  gemm(ffh16, 1024, w_ff2, 1024, ff_b2, ffo_f, nullptr, (int)M, 256, 1024, 0);
  residual_ln<<<dim3((unsigned)M), dim3(256), 0, stream>>>(
      ffo_f, q2_f, ln3_g, ln3_b, (float*)d_out, nullptr);
}
